// SingleLayerSNN_52845277610451
// MI455X (gfx1250) — matrix operand order repaired and checked by reading, not compiled, not run on hardware
//
#include <hip/hip_runtime.h>
#include <hip/hip_bf16.h>

typedef __attribute__((ext_vector_type(16))) _Float16 v16h;
typedef __attribute__((ext_vector_type(8)))  float    v8f;

// Problem constants (from reference setup_inputs)
#define BATCH   128
#define TSTEPS  1000
#define FDIM    2048
#define CDIM    10
#define NPAD    16
#define MROWS   (BATCH * TSTEPS)      // 128000
#define KCHUNKS (FDIM / 32)           // 64
#define MTILES  4                     // M-tiles (16 rows each) per wave
#define WAVES   8                     // waves per block (256 threads, wave32)
#define ROWS_PER_BLOCK (WAVES * MTILES * 16)   // 512
#define GEMM_BLOCKS (MROWS / ROWS_PER_BLOCK)   // 250

#define V_TH    0.4f
#define REFRAC  3.0f

// ---------------------------------------------------------------------------
// Phase 1: I[m, n] = x[m, :] @ Wpad[:, n]   (m = b*T + t, n = 0..15, cols >=10 are 0)
// f16 WMMA, f32 accumulate. x streamed once from HBM (the roofline cost).
// ---------------------------------------------------------------------------
__global__ __launch_bounds__(256) void snn_gemm_wmma(
    const float* __restrict__ x,      // [MROWS, FDIM] fp32 (binary 0/1)
    const float* __restrict__ W,      // [FDIM, CDIM] fp32
    float* __restrict__ I)            // [MROWS, NPAD] fp32 out
{
    // LDS W in f16, laid out [k][n] with n contiguous (32B per k row); B fragment gathered per lane:
    // lane n = mrow reads column n, one half from each of its 16 k rows (stride NPAD).
    __shared__ _Float16 sW[FDIM * NPAD];   // 64 KB

    const int tid = threadIdx.x;

    // Cooperative convert W (2048x10 fp32) -> sW (2048x16 f16), zero-padded.
    #pragma unroll
    for (int i = 0; i < FDIM / 256; ++i) {      // 8 rows per thread
        const int k = i * 256 + tid;
        const float* wr = W + k * CDIM;
        _Float16* dst = sW + k * NPAD;
        #pragma unroll
        for (int n = 0; n < NPAD; ++n)
            dst[n] = (n < CDIM) ? (_Float16)wr[n] : (_Float16)0.0f;
    }
    __syncthreads();

    const int lane = tid & 31;
    const int wave = tid >> 5;
    const int mrow = lane & 15;        // A-row within tile / D column (N index)
    const int hi   = lane >> 4;        // lane group: 0 or 1
    const int koff = hi * 8;           // A fragment K sub-offset per ISA layout

    const long rowBase = (long)(blockIdx.x * WAVES + wave) * (MTILES * 16);

    const float* xrow[MTILES];
    #pragma unroll
    for (int mt = 0; mt < MTILES; ++mt)
        xrow[mt] = x + (rowBase + mt * 16 + mrow) * (long)FDIM;

    v8f acc[MTILES];
    #pragma unroll
    for (int mt = 0; mt < MTILES; ++mt) acc[mt] = (v8f){};

    for (int kc = 0; kc < KCHUNKS; ++kc) {
        const int kb = kc * 32;

        // B fragment: lane holds column n = mrow; halves 0..7 <- K = kb + koff .. +7, halves 8..15 <- K = kb + 16 + koff .. +7.
        v16h bfrag;
        #pragma unroll
        for (int s = 0; s < 16; ++s)
            bfrag[s] = sW[(kb + koff + (s & 7) + (s >> 3) * 16) * NPAD + mrow];

        v16h af[MTILES];
        #pragma unroll
        for (int mt = 0; mt < MTILES; ++mt) {
            const float* p0 = xrow[mt] + kb + koff;
            // halves 0..7  <- K = kb + koff .. +7
            float4 a0 = *(const float4*)(p0);
            float4 a1 = *(const float4*)(p0 + 4);
            // halves 8..15 <- K = kb + 16 + koff .. +7
            float4 a2 = *(const float4*)(p0 + 16);
            float4 a3 = *(const float4*)(p0 + 20);
            if (kc + 1 < KCHUNKS)
                __builtin_prefetch(p0 + 32, 0, 1);   // next K-chunk cacheline

            v16h a;
            a[0]  = (_Float16)a0.x; a[1]  = (_Float16)a0.y;
            a[2]  = (_Float16)a0.z; a[3]  = (_Float16)a0.w;
            a[4]  = (_Float16)a1.x; a[5]  = (_Float16)a1.y;
            a[6]  = (_Float16)a1.z; a[7]  = (_Float16)a1.w;
            a[8]  = (_Float16)a2.x; a[9]  = (_Float16)a2.y;
            a[10] = (_Float16)a2.z; a[11] = (_Float16)a2.w;
            a[12] = (_Float16)a3.x; a[13] = (_Float16)a3.y;
            a[14] = (_Float16)a3.z; a[15] = (_Float16)a3.w;
            af[mt] = a;
        }

        #pragma unroll
        for (int mt = 0; mt < MTILES; ++mt) {
            acc[mt] = __builtin_amdgcn_wmma_f32_16x16x32_f16(
                false, af[mt], false, bfrag, (short)0, acc[mt], false, false);
        }
    }

    // Store D: lane n = mrow; VGPR r -> row (hi*8 + r). Coalesced 64B segments.
    #pragma unroll
    for (int mt = 0; mt < MTILES; ++mt) {
        const long tbase = rowBase + mt * 16;
        #pragma unroll
        for (int r = 0; r < 8; ++r) {
            I[(tbase + hi * 8 + r) * NPAD + mrow] = acc[mt][r];
        }
    }
}

// ---------------------------------------------------------------------------
// Phase 2: per-(b,c) LIF recurrence over T steps; 1280 independent threads.
// Reads I from L2-resident workspace (~5 MB), writes spike counts [128,10].
// ---------------------------------------------------------------------------
__global__ __launch_bounds__(256) void snn_lif_scan(
    const float* __restrict__ I,      // [MROWS, NPAD]
    float* __restrict__ out)          // [BATCH, CDIM]
{
    const int idx = blockIdx.x * blockDim.x + threadIdx.x;
    if (idx >= BATCH * CDIM) return;
    const int b = idx / CDIM;
    const int c = idx % CDIM;

    const float* p = I + (long)b * TSTEPS * NPAD + c;

    float V = 0.0f, refr = 0.0f, cnt = 0.0f;
    for (int t = 0; t < TSTEPS; ++t) {
        const float It = p[(long)t * NPAD];
        const float active = (refr <= 0.0f) ? 1.0f : 0.0f;
        V += It * active;
        const float s = ((V - V_TH) > 0.0f) ? active : 0.0f;  // spike * active
        const bool fired = (s > 0.0f);
        V    = fired ? 0.0f   : V;
        cnt += s;
        refr = fired ? REFRAC : refr;
        refr = fmaxf(refr - 1.0f, 0.0f);
    }
    out[idx] = cnt;
}

// ---------------------------------------------------------------------------
extern "C" void kernel_launch(void* const* d_in, const int* in_sizes, int n_in,
                              void* d_out, int out_size, void* d_ws, size_t ws_size,
                              hipStream_t stream) {
    const float* x = (const float*)d_in[0];   // [128,1000,2048] fp32
    const float* W = (const float*)d_in[1];   // [2048,10] fp32
    float* I = (float*)d_ws;                  // needs MROWS*NPAD*4 = 8,192,000 B

    snn_gemm_wmma<<<GEMM_BLOCKS, 256, 0, stream>>>(x, W, I);

    const int nseq = BATCH * CDIM;            // 1280
    snn_lif_scan<<<(nseq + 255) / 256, 256, 0, stream>>>(I, (float*)d_out);
}
